// SAINT_86801289052599
// MI455X (gfx1250) — compile-verified, matrix operand order checked by reading, not run on hardware
//
#include <hip/hip_runtime.h>
#include <hip/hip_bf16.h>
#include <math.h>

// ---------------------------------------------------------------------------
// Fused SAINT-style transformer forward for MI455X (gfx1250, wave32, WMMA).
// One workgroup (8 waves / 256 threads) processes 4 sequences (128 padded
// rows).  Residual stream h stays in LDS (f16) across all 4 layers.  All
// GEMMs run on v_wmma_f32_16x16x32_f16 with f32 accumulation.  Weights are
// pre-converted to f16 and transposed ([N][K]) into the workspace so B
// fragments stream from L2 with global_load_b128.  Each wave owns a fixed
// 16-row slab for the large GEMMs so A fragments are hoisted into registers
// and reused across all N tiles.
// ---------------------------------------------------------------------------

typedef __attribute__((ext_vector_type(16))) _Float16 v16h;
typedef __attribute__((ext_vector_type(8)))  _Float16 v8h;
typedef __attribute__((ext_vector_type(8)))  float    v8f;

namespace cfg {
constexpr int Bsz = 4096, S = 30, D = 320, H = 5, DH = 64, L = 4, DFF = 1280, FIN = 80;
constexpr int BPB = 4;           // batch elements per workgroup
constexpr int ROWS = 128;        // 4 seqs * 32 padded rows
// LDS strides (in elements); all chosen so fragment loads stay 16B aligned
constexpr int HS = 328;          // h (f16)
constexpr int CS = 321;          // f32 accumulation buffer
constexpr int QS = 72;           // q / headtile (f16)
constexpr int VS = 32;           // v-transposed (f16) per seq [64][32]
constexpr int SS = 33;           // scores (f32)
constexpr int PS = 32;           // softmax probs (f16)
constexpr int GS = 264;          // FFN G chunk per wave [16][264] (f16)
constexpr int XS = 104;          // input x staging (f16)
// LDS byte offsets
constexpr int OFF_H = 0;
constexpr int OFF_C = OFF_H + ROWS * HS * 2;                 // 83968
constexpr int OFF_A = OFF_C + ROWS * CS * 4;                 // 248320 (union region)
constexpr int OFF_Q = OFF_A;
constexpr int OFF_K = OFF_Q + ROWS * QS * 2;
constexpr int OFF_V = OFF_K + ROWS * QS * 2;
constexpr int OFF_S = OFF_V + BPB * DH * VS * 2;
constexpr int OFF_P = OFF_S + ROWS * SS * 4;
constexpr int LDS_BYTES = OFF_P + ROWS * PS * 2;             // 326656 <= 320KB
// workspace (f16 element offsets)
constexpr size_t O_INW = 0;
constexpr size_t SZ_INW = 320 * 96;
constexpr size_t O_WQ = O_INW + SZ_INW;
constexpr size_t SZ_HEADW = (size_t)L * H * DH * DH;
constexpr size_t O_WK = O_WQ + SZ_HEADW;
constexpr size_t O_WV = O_WK + SZ_HEADW;
constexpr size_t O_WO = O_WV + SZ_HEADW;
constexpr size_t O_W1 = O_WO + (size_t)L * D * D;
constexpr size_t O_W2 = O_W1 + (size_t)L * D * DFF;
constexpr size_t WS_HALVES = O_W2 + (size_t)L * DFF * D;     // ~7.6 MiB
} // namespace cfg

// --- WMMA wrapper -----------------------------------------------------------
__device__ __forceinline__ v8f wmma_f16(v16h a, v16h b, v8f c) {
  return __builtin_amdgcn_wmma_f32_16x16x32_f16(false, a, false, b, (short)0, c,
                                                false, false);
}

// 16x32 f16 fragment, A-style layout (ISA 7.12.2): lane m = lane&15,
// kbase = (lane>>4)*8; elements 0..7 -> k = kbase+0..7, 8..15 -> k = 16+kbase+0..7.
// Works for both LDS (ds_load_b128) and global (global_load_b128) pointers.
// B fragments are loaded with the same pattern from B^T stored [N][K].
__device__ __forceinline__ v16h load_frag(const _Float16* __restrict__ base,
                                          int stride, int m0, int k0) {
  const int lane = threadIdx.x & 31;
  const _Float16* p =
      base + (size_t)(m0 + (lane & 15)) * stride + (k0 + ((lane >> 4) << 3));
  union { v16h v; v8h h[2]; } u;
  u.h[0] = *(const v8h*)p;
  u.h[1] = *(const v8h*)(p + 16);
  return u.v;
}

// --- weight convert + transpose: dst[n][k] = src[k][n] (f32->f16, zero pad) --
__global__ void conv_t_kernel(const float* __restrict__ src,
                              _Float16* __restrict__ dst, int K, int N, int Kpad) {
  int idx = blockIdx.x * blockDim.x + threadIdx.x;
  if (idx >= N * Kpad) return;
  int n = idx / Kpad, k = idx - n * Kpad;
  float v = (k < K) ? src[(size_t)k * N + n] : 0.f;
  dst[idx] = (_Float16)v;
}

extern __shared__ __align__(16) char smem[];

// --- LayerNorm over 320 cols: h = LN(cbuf + bias + h) ------------------------
__device__ __forceinline__ void ln_block(float* __restrict__ cbuf,
                                         _Float16* __restrict__ hbuf,
                                         const float* __restrict__ bias,
                                         const float* __restrict__ gam,
                                         const float* __restrict__ bet,
                                         int wave, int lane) {
  using namespace cfg;
  for (int i = 0; i < 16; ++i) {
    int m = wave * 16 + i;
    float v[10];
    float s = 0.f, s2 = 0.f;
#pragma unroll
    for (int j = 0; j < 10; ++j) {
      int col = lane + j * 32;
      float val = cbuf[m * CS + col] + bias[col] + (float)hbuf[m * HS + col];
      v[j] = val; s += val; s2 += val * val;
    }
#pragma unroll
    for (int off = 16; off > 0; off >>= 1) {
      s += __shfl_xor(s, off, 32);
      s2 += __shfl_xor(s2, off, 32);
    }
    float mean = s * (1.f / 320.f);
    float var = fmaxf(s2 * (1.f / 320.f) - mean * mean, 0.f);
    float inv = rsqrtf(var + 1e-5f);
#pragma unroll
    for (int j = 0; j < 10; ++j) {
      int col = lane + j * 32;
      float nv = (v[j] - mean) * inv * gam[col] + bet[col];
      hbuf[m * HS + col] = (_Float16)nv;
    }
  }
}

__global__ __launch_bounds__(256, 1) void saint_fused(
    const _Float16* __restrict__ ws, const float* __restrict__ x,
    const float* __restrict__ in_b, const float* __restrict__ bq,
    const float* __restrict__ bk, const float* __restrict__ bv,
    const float* __restrict__ bo, const float* __restrict__ ln1g,
    const float* __restrict__ ln1b, const float* __restrict__ b1,
    const float* __restrict__ b2, const float* __restrict__ ln2g,
    const float* __restrict__ ln2b, const float* __restrict__ poolW,
    const float* __restrict__ poolb, const float* __restrict__ clsW1,
    const float* __restrict__ clsb1, const float* __restrict__ clsW2,
    const float* __restrict__ clsb2, float* __restrict__ out) {
  using namespace cfg;
  _Float16* hbuf = (_Float16*)(smem + OFF_H);
  float*    cbuf = (float*)(smem + OFF_C);
  _Float16* qbuf = (_Float16*)(smem + OFF_Q);   // Q, then head tile
  _Float16* kbuf = (_Float16*)(smem + OFF_K);
  _Float16* vtb  = (_Float16*)(smem + OFF_V);   // [seq][d=64][kk=32]
  float*    sbuf = (float*)(smem + OFF_S);
  _Float16* pbuf = (_Float16*)(smem + OFF_P);
  _Float16* xbuf = (_Float16*)(smem + OFF_A);   // aliases attention region
  _Float16* gbufA = (_Float16*)(smem + OFF_A);  // aliases attention region

  const int tid = threadIdx.x;
  const int lane = tid & 31, wave = tid >> 5;
  const int cn = lane & 15, chh = lane >> 4;    // C-layout: col = n0+cn, row = m0+8*chh+r
  const int b0 = blockIdx.x * BPB;
  const int mw = wave * 16;                     // this wave's 16-row slab

  // ---- stage x -> xbuf (f16, K padded 80->96, padded rows zeroed) ----
  for (int idx = tid; idx < ROWS * 96; idx += 256) {
    int m = idx / 96, k = idx - (idx / 96) * 96;
    int seq = m >> 5, s = m & 31;
    float v = 0.f;
    if (s < S && k < FIN) v = x[((size_t)(b0 + seq) * S + s) * FIN + k];
    xbuf[m * XS + k] = (_Float16)v;
  }
  __syncthreads();

  // ---- input projection + positional encoding -> hbuf ----
  {
    const _Float16* inWt = ws + O_INW;
    v16h a0 = load_frag(xbuf, XS, mw, 0);
    v16h a1 = load_frag(xbuf, XS, mw, 32);
    v16h a2 = load_frag(xbuf, XS, mw, 64);
    for (int nt = 0; nt < 20; ++nt) {
      int n0 = nt * 16;
      v8f acc = {};
      acc = wmma_f16(a0, load_frag(inWt, 96, n0, 0), acc);
      acc = wmma_f16(a1, load_frag(inWt, 96, n0, 32), acc);
      acc = wmma_f16(a2, load_frag(inWt, 96, n0, 64), acc);
      union { v8f v; float f[8]; } u; u.v = acc;
      int n = n0 + cn;
      float bias = in_b[n];
      float freq = __expf(-(float)(n & ~1) * (9.210340372f / 320.f));
#pragma unroll
      for (int r = 0; r < 8; ++r) {
        int m = mw + chh * 8 + r;
        int s = m & 31;
        float sv, cv;
        __sincosf((float)s * freq, &sv, &cv);
        float pe = (n & 1) ? cv : sv;
        float v = u.f[r] + bias + pe;
        if (s >= S) v = 0.f;  // keep padded rows finite & zero
        hbuf[m * HS + n] = (_Float16)v;
      }
    }
  }
  __syncthreads();

  // ============================ layer loop ============================
  for (int l = 0; l < L; ++l) {
    const _Float16* wq = ws + O_WQ + (size_t)l * H * DH * DH;
    const _Float16* wk = ws + O_WK + (size_t)l * H * DH * DH;
    const _Float16* wv = ws + O_WV + (size_t)l * H * DH * DH;
    const _Float16* wo = ws + O_WO + (size_t)l * D * D;
    const _Float16* w1 = ws + O_W1 + (size_t)l * D * DFF;
    const _Float16* w2 = ws + O_W2 + (size_t)l * DFF * D;

    for (int hd = 0; hd < H; ++hd) {
      // --- Q,K,V for this head: A = hbuf[:, 64*hd:...] (wave's 16 rows) ---
      {
        v16h a0 = load_frag(hbuf + hd * DH, HS, mw, 0);
        v16h a1 = load_frag(hbuf + hd * DH, HS, mw, 32);
        for (int mat = 0; mat < 3; ++mat) {
          const _Float16* wsel = (mat == 0) ? wq : (mat == 1) ? wk : wv;
          const float* bsel = (mat == 0) ? bq : (mat == 1) ? bk : bv;
          wsel += (size_t)hd * DH * DH;
          bsel += ((size_t)l * H + hd) * DH;
          for (int nt = 0; nt < 4; ++nt) {
            int n0 = nt * 16;
            v8f acc = {};
            acc = wmma_f16(a0, load_frag(wsel, DH, n0, 0), acc);
            acc = wmma_f16(a1, load_frag(wsel, DH, n0, 32), acc);
            union { v8f v; float f[8]; } u; u.v = acc;
            int n = n0 + cn;
            float bias = bsel[n];
#pragma unroll
            for (int r = 0; r < 8; ++r) {
              int m = mw + chh * 8 + r;
              float v = u.f[r] + bias;
              if (mat == 0) qbuf[m * QS + n] = (_Float16)v;
              else if (mat == 1) kbuf[m * QS + n] = (_Float16)v;
              else { // V stored transposed per sequence: vt[seq][d][kk]
                int seq = m >> 5, kk = m & 31;
                vtb[((size_t)seq * DH + n) * VS + kk] = (_Float16)v;
              }
            }
          }
        }
      }
      __syncthreads();

      // --- scores = Q K^T / 8 (per sequence, 32x32, K=64) ---
      for (int job = wave; job < 16; job += 8) {
        int seq = job >> 2, t = job & 3;
        int qt = t >> 1, kt2 = t & 1;
        int m0 = seq * 32 + qt * 16, n0 = kt2 * 16;
        v8f acc = {};
#pragma unroll
        for (int kt = 0; kt < 2; ++kt) {
          v16h a = load_frag(qbuf, QS, m0, kt * 32);
          v16h b = load_frag(kbuf + (size_t)seq * 32 * QS, QS, n0, kt * 32);
          acc = wmma_f16(a, b, acc);
        }
        union { v8f v; float f[8]; } u; u.v = acc;
        int kk = n0 + cn;
#pragma unroll
        for (int r = 0; r < 8; ++r) {
          int m = m0 + chh * 8 + r;
          float v = u.f[r] * 0.125f;
          if (kk >= S) v = -1e30f;  // mask padded keys
          sbuf[m * SS + kk] = v;
        }
      }
      __syncthreads();

      // --- softmax per query row ---
      if (tid < ROWS) {
        int m = tid;
        float mx = -3.0e38f;
        for (int kk = 0; kk < 32; ++kk) mx = fmaxf(mx, sbuf[m * SS + kk]);
        float sum = 0.f;
        for (int kk = 0; kk < 32; ++kk) sum += __expf(sbuf[m * SS + kk] - mx);
        float inv = 1.f / sum;
        for (int kk = 0; kk < 32; ++kk)
          pbuf[m * PS + kk] = (_Float16)(__expf(sbuf[m * SS + kk] - mx) * inv);
      }
      __syncthreads();

      // --- head = P @ V (per sequence, 32x64, K=32) -> reuse qbuf ---
      for (int job = wave; job < 32; job += 8) {
        int seq = job >> 3, t = job & 7;
        int mt = t >> 2, nt = t & 3;
        int m0 = seq * 32 + mt * 16, n0 = nt * 16;
        v16h a = load_frag(pbuf, PS, m0, 0);
        v16h b = load_frag(vtb + (size_t)seq * DH * VS, VS, n0, 0);
        v8f acc = {};
        acc = wmma_f16(a, b, acc);
        union { v8f v; float f[8]; } u; u.v = acc;
        int n = n0 + cn;
#pragma unroll
        for (int r = 0; r < 8; ++r)
          qbuf[(m0 + chh * 8 + r) * QS + n] = (_Float16)u.f[r];
      }
      __syncthreads();

      // --- attn_out += head @ Wo[64*hd : 64*hd+64, :] -> cbuf (f32) ---
      {
        v16h a0 = load_frag(qbuf, QS, mw, 0);
        v16h a1 = load_frag(qbuf, QS, mw, 32);
        for (int nt = 0; nt < 20; ++nt) {
          int n0 = nt * 16;
          v8f acc = {};
          acc = wmma_f16(a0, load_frag(wo, D, n0, hd * DH), acc);
          acc = wmma_f16(a1, load_frag(wo, D, n0, hd * DH + 32), acc);
          union { v8f v; float f[8]; } u; u.v = acc;
          int n = n0 + cn;
#pragma unroll
          for (int r = 0; r < 8; ++r) {
            int m = mw + chh * 8 + r;
            float prev = (hd == 0) ? 0.f : cbuf[m * CS + n];
            cbuf[m * CS + n] = prev + u.f[r];
          }
        }
      }
      __syncthreads();
    } // head loop

    // --- h = LN1(h + attn_out + bo) ---
    ln_block(cbuf, hbuf, bo + (size_t)l * D, ln1g + (size_t)l * D,
             ln1b + (size_t)l * D, wave, lane);
    __syncthreads();

    // --- FFN, chunked over DFF (5 chunks of 256); each wave owns 16 rows ---
    {
      _Float16* gb = gbufA + (size_t)wave * 16 * GS;
      // hoist the wave's A fragments of h (K = 320 -> 10 fragments)
      v16h af[10];
#pragma unroll
      for (int kt = 0; kt < 10; ++kt)
        af[kt] = load_frag(hbuf, HS, mw, kt * 32);

      for (int chunk = 0; chunk < 5; ++chunk) {
        // G = relu(h @ W1[:, chunk] + b1)   (16 rows x 256 per wave)
        for (int nt = 0; nt < 16; ++nt) {
          int n0g = chunk * 256 + nt * 16;
          v8f acc = {};
#pragma unroll
          for (int kt = 0; kt < 10; ++kt)
            acc = wmma_f16(af[kt], load_frag(w1, D, n0g, kt * 32), acc);
          union { v8f v; float f[8]; } u; u.v = acc;
          float bias = b1[(size_t)l * DFF + n0g + cn];
#pragma unroll
          for (int r = 0; r < 8; ++r) {
            float v = fmaxf(u.f[r] + bias, 0.f);
            gb[(chh * 8 + r) * GS + nt * 16 + cn] = (_Float16)v;
          }
        }
        // wave-private LDS RAW: G stores -> G fragment loads
        asm volatile("s_wait_dscnt 0x0" ::: "memory");

        // hoist G fragments (K = 256 -> 8 fragments), reuse across 20 N tiles
        v16h ag[8];
#pragma unroll
        for (int kt = 0; kt < 8; ++kt)
          ag[kt] = load_frag(gb, GS, 0, kt * 32);

        // cbuf += G @ W2[chunk, :]   (16 rows x 320 per wave, K=256)
        for (int nt = 0; nt < 20; ++nt) {
          union { v8f v; float f[8]; } u;
          int n = nt * 16 + cn;
#pragma unroll
          for (int r = 0; r < 8; ++r)
            u.f[r] = (chunk == 0) ? 0.f
                                  : cbuf[(mw + chh * 8 + r) * CS + n];
          v8f acc = u.v;
#pragma unroll
          for (int kt = 0; kt < 8; ++kt)
            acc = wmma_f16(ag[kt],
                           load_frag(w2, DFF, nt * 16, chunk * 256 + kt * 32),
                           acc);
          u.v = acc;
#pragma unroll
          for (int r = 0; r < 8; ++r)
            cbuf[(mw + chh * 8 + r) * CS + n] = u.f[r];
        }
      }
    }
    __syncthreads();

    // --- h = LN2(h + ffn + b2) ---
    ln_block(cbuf, hbuf, b2 + (size_t)l * D, ln2g + (size_t)l * D,
             ln2b + (size_t)l * D, wave, lane);
    __syncthreads();
  } // layer loop

  // ============== attention pooling + classifier (small, scalar) ==============
  float* pscore = (float*)(smem + OFF_A);
  float* betab = pscore + 128;
  float* zbuf = betab + 128;        // [4][320]
  float* hid = zbuf + BPB * D;      // [4][160]
  if (tid < ROWS) {
    int m = tid, s = m & 31;
    float acc = poolb[0];
    for (int j = 0; j < D; ++j) acc += (float)hbuf[m * HS + j] * poolW[j];
    pscore[m] = (s < S) ? acc : -1e30f;
  }
  __syncthreads();
  if (tid < BPB) {
    int seq = tid;
    float mx = -3.0e38f;
    for (int s = 0; s < S; ++s) mx = fmaxf(mx, pscore[seq * 32 + s]);
    float sum = 0.f;
    for (int s = 0; s < S; ++s) sum += __expf(pscore[seq * 32 + s] - mx);
    float inv = 1.f / sum;
    for (int s = 0; s < S; ++s)
      betab[seq * 32 + s] = __expf(pscore[seq * 32 + s] - mx) * inv;
  }
  __syncthreads();
  for (int idx = tid; idx < BPB * D; idx += 256) {
    int seq = idx / D, d = idx - seq * D;
    float acc = 0.f;
    for (int s = 0; s < S; ++s)
      acc += betab[seq * 32 + s] * (float)hbuf[(seq * 32 + s) * HS + d];
    zbuf[idx] = acc;
  }
  __syncthreads();
  for (int idx = tid; idx < BPB * 160; idx += 256) {
    int seq = idx / 160, j = idx - seq * 160;
    float acc = clsb1[j];
    for (int d = 0; d < D; ++d) acc += zbuf[seq * D + d] * clsW1[d * 160 + j];
    hid[idx] = fmaxf(acc, 0.f);
  }
  __syncthreads();
  if (tid < BPB) {
    int seq = tid;
    float acc = clsb2[0];
    for (int j = 0; j < 160; ++j) acc += hid[seq * 160 + j] * clsW2[j];
    out[b0 + seq] = acc;                               // logits
    out[Bsz + b0 + seq] = 1.f / (1.f + expf(-acc));    // probs
  }
}

// ---------------------------------------------------------------------------
extern "C" void kernel_launch(void* const* d_in, const int* in_sizes, int n_in,
                              void* d_out, int out_size, void* d_ws,
                              size_t ws_size, hipStream_t stream) {
  using namespace cfg;
  const float* x = (const float*)d_in[0];
  const float* in_W = (const float*)d_in[1];
  const float* in_b = (const float*)d_in[2];
  const float* Wq = (const float*)d_in[3];
  const float* bq = (const float*)d_in[4];
  const float* Wk = (const float*)d_in[5];
  const float* bk = (const float*)d_in[6];
  const float* Wv = (const float*)d_in[7];
  const float* bv = (const float*)d_in[8];
  const float* Wo = (const float*)d_in[9];
  const float* bo = (const float*)d_in[10];
  const float* ln1g = (const float*)d_in[11];
  const float* ln1b = (const float*)d_in[12];
  const float* W1 = (const float*)d_in[13];
  const float* b1 = (const float*)d_in[14];
  const float* W2 = (const float*)d_in[15];
  const float* b2 = (const float*)d_in[16];
  const float* ln2g = (const float*)d_in[17];
  const float* ln2b = (const float*)d_in[18];
  const float* poolW = (const float*)d_in[19];
  const float* poolb = (const float*)d_in[20];
  const float* clsW1 = (const float*)d_in[21];
  const float* clsb1 = (const float*)d_in[22];
  const float* clsW2 = (const float*)d_in[23];
  const float* clsb2 = (const float*)d_in[24];
  float* out = (float*)d_out;
  _Float16* ws = (_Float16*)d_ws;

  auto conv = [&](const float* src, _Float16* dst, int K, int N, int Kpad) {
    int total = N * Kpad;
    conv_t_kernel<<<(total + 255) / 256, 256, 0, stream>>>(src, dst, K, N, Kpad);
  };

  // weight conversion + transpose into workspace (f16, [N][K])
  conv(in_W, ws + O_INW, FIN, D, 96);
  for (int l = 0; l < L; ++l)
    for (int h = 0; h < H; ++h) {
      size_t o = ((size_t)l * H + h) * DH * DH;
      conv(Wq + o, ws + O_WQ + o, DH, DH, DH);
      conv(Wk + o, ws + O_WK + o, DH, DH, DH);
      conv(Wv + o, ws + O_WV + o, DH, DH, DH);
    }
  for (int l = 0; l < L; ++l) {
    conv(Wo + (size_t)l * D * D, ws + O_WO + (size_t)l * D * D, D, D, D);
    conv(W1 + (size_t)l * D * DFF, ws + O_W1 + (size_t)l * D * DFF, D, DFF, D);
    conv(W2 + (size_t)l * DFF * D, ws + O_W2 + (size_t)l * DFF * D, DFF, D, DFF);
  }

  (void)hipFuncSetAttribute(reinterpret_cast<const void*>(&saint_fused),
                            hipFuncAttributeMaxDynamicSharedMemorySize,
                            LDS_BYTES);

  saint_fused<<<Bsz / BPB, 256, LDS_BYTES, stream>>>(
      ws, x, in_b, bq, bk, bv, bo, ln1g, ln1b, b1, b2, ln2g, ln2b, poolW,
      poolb, clsW1, clsb1, clsW2, clsb2, out);
}
